// SelfAttention_15994458211390
// MI455X (gfx1250) — matrix operand order repaired and checked by reading, not compiled, not run on hardware
//
#include <hip/hip_runtime.h>

typedef __attribute__((ext_vector_type(16))) __bf16       v16bf;
typedef __attribute__((ext_vector_type(8)))  float        v8f;
typedef __attribute__((ext_vector_type(4)))  unsigned int v4u;

union FragU { v16bf bf; v4u u[2]; };
union Pack8 { v4u u; unsigned int w[4]; };

#define NB   8      // batch
#define CC   512    // channels
#define HN   8      // heads
#define HC   64     // head channels
#define NPIX 4096   // h*w

__device__ __forceinline__ unsigned short f32_to_bf16(float f) {
  unsigned int u = __builtin_bit_cast(unsigned int, f);
  u += 0x7FFFu + ((u >> 16) & 1u);   // round-to-nearest-even
  return (unsigned short)(u >> 16);
}

__device__ __forceinline__ v8f wmma_bf16(v16bf a, v16bf b, v8f c) {
  // D = A(16x32) * B(32x16) + C, f32 accum
  return __builtin_amdgcn_wmma_f32_16x16x32_bf16(false, a, false, b, (short)0, c,
                                                 false, false);
}

// ---------------------------------------------------------------------------
// x [b, c, n] f32  ->  xt [b, n, c] bf16   (32x32 LDS transpose tiles)
// ---------------------------------------------------------------------------
__global__ __launch_bounds__(256) void k_transpose_cvt(const float* __restrict__ x,
                                                       unsigned short* __restrict__ xt) {
  __shared__ float tile[32][33];
  const int n0 = blockIdx.x * 32, c0 = blockIdx.y * 32, b = blockIdx.z;
  const int tx = threadIdx.x, ty = threadIdx.y;     // 32 x 8
  const float* xb = x + (size_t)b * CC * NPIX;
  for (int j = 0; j < 4; ++j) {
    int c = c0 + ty + j * 8;
    tile[ty + j * 8][tx] = xb[(size_t)c * NPIX + n0 + tx];
  }
  __syncthreads();
  unsigned short* xtb = xt + (size_t)b * NPIX * CC;
  for (int j = 0; j < 4; ++j) {
    int n = n0 + ty + j * 8;
    xtb[(size_t)n * CC + c0 + tx] = f32_to_bf16(tile[tx][ty + j * 8]);
  }
}

// ---------------------------------------------------------------------------
// elementwise f32 -> bf16 (weights)
// ---------------------------------------------------------------------------
__global__ __launch_bounds__(256) void k_cvt(const float* __restrict__ s,
                                             unsigned short* __restrict__ d, int n) {
  int i = blockIdx.x * 256 + threadIdx.x;
  if (i < n) d[i] = f32_to_bf16(s[i]);
}

// ---------------------------------------------------------------------------
// GEMM: Out[m,n] = sum_k A[m,k] * Bt[n,k] + bias[m]
//   A  : [512, 512] bf16, K contiguous
//   Bt : [batch][4096][512] bf16, K contiguous per n-row
// MODE 0: store bf16 [b][m][n]        (Q, K projections)
// MODE 1: store bf16 transposed per head: vt[b][h][n][d]   (V projection)
// MODE 2: store f32  [b][m][n]        (final projection -> d_out)
// Block tile 128x128, 8 waves, wave tile 32x64.
// Double-buffered LDS, one barrier per 32-wide K-step; register prefetch of
// tile k+1 overlaps the 8 WMMAs of tile k; global_prefetch_b8 two tiles ahead.
// ---------------------------------------------------------------------------
template <int MODE>
__global__ __launch_bounds__(256) void k_gemm(const unsigned short* __restrict__ A,
                                              const unsigned short* __restrict__ Bt,
                                              const float* __restrict__ bias,
                                              void* __restrict__ Out) {
  __shared__ unsigned short ldsA[2][128 * 40];  // 32 bf16 + 8 pad per row (80B rows)
  __shared__ unsigned short ldsB[2][128 * 40];
  const int t = threadIdx.x;
  const int lane = t & 31;
  const int w = t >> 5;
  const int hi = (lane >> 4) & 1, l15 = lane & 15;
  const int wm = w >> 1, wn = w & 1;
  const int n0 = blockIdx.x * 128, m0 = blockIdx.y * 128, b = blockIdx.z;
  const unsigned short* Bb = Bt + (size_t)b * NPIX * CC;

  // staging: each thread owns two (row, chunk) slots of the 128x32 tile
  const int row0 = t >> 2, ch = t & 3;
  const unsigned short* aSrc0 = A + (size_t)(m0 + row0) * CC + ch * 8;
  const unsigned short* aSrc1 = aSrc0 + (size_t)64 * CC;
  const unsigned short* bSrc0 = Bb + (size_t)(n0 + row0) * CC + ch * 8;
  const unsigned short* bSrc1 = bSrc0 + (size_t)64 * CC;
  const int st0 = row0 * 40 + ch * 8;
  const int st1 = st0 + 64 * 40;

  v8f acc[2][4] = {};

  // preload K-tile 0
  v4u pa0 = *(const v4u*)(aSrc0), pa1 = *(const v4u*)(aSrc1);
  v4u pb0 = *(const v4u*)(bSrc0), pb1 = *(const v4u*)(bSrc1);
  *(v4u*)(ldsA[0] + st0) = pa0;
  *(v4u*)(ldsA[0] + st1) = pa1;
  *(v4u*)(ldsB[0] + st0) = pb0;
  *(v4u*)(ldsB[0] + st1) = pb1;
  __syncthreads();

  const int NT = CC / 32;  // 16 K-steps
  for (int kt = 0; kt < NT; ++kt) {
    const int cur = kt & 1;
    if (kt + 1 < NT) {
      const int kk = (kt + 1) * 32;
      pa0 = *(const v4u*)(aSrc0 + kk);
      pa1 = *(const v4u*)(aSrc1 + kk);
      pb0 = *(const v4u*)(bSrc0 + kk);
      pb1 = *(const v4u*)(bSrc1 + kk);
      if (kt + 2 < NT) {  // pull lines toward the WGP two tiles ahead
        __builtin_prefetch(aSrc0 + kk + 32, 0, 0);
        __builtin_prefetch(bSrc0 + kk + 32, 0, 0);
      }
    }

    v16bf afrag[2], bfrag[4];
    for (int i = 0; i < 2; ++i) {
      const unsigned short* p = ldsA[cur] + (wm * 32 + i * 16 + l15) * 40;
      FragU f;
      f.u[0] = *(const v4u*)(p + hi * 16);       // K = kk + hi*16 .. +7
      f.u[1] = *(const v4u*)(p + hi * 16 + 8);   // .. +15
      afrag[i] = f.bf;
    }
    for (int j = 0; j < 4; ++j) {
      const unsigned short* p = ldsB[cur] + (wn * 64 + j * 16 + l15) * 40 + hi * 16;
      FragU f;
      f.u[0] = *(const v4u*)(p);                 // K = kk + hi*16 .. +7
      f.u[1] = *(const v4u*)(p + 8);             // .. +15
      bfrag[j] = f.bf;
    }
    for (int i = 0; i < 2; ++i)
      for (int j = 0; j < 4; ++j)
        acc[i][j] = wmma_bf16(afrag[i], bfrag[j], acc[i][j]);

    if (kt + 1 < NT) {
      // buffer cur^1 was last read in iteration kt-1 (before previous barrier)
      *(v4u*)(ldsA[cur ^ 1] + st0) = pa0;
      *(v4u*)(ldsA[cur ^ 1] + st1) = pa1;
      *(v4u*)(ldsB[cur ^ 1] + st0) = pb0;
      *(v4u*)(ldsB[cur ^ 1] + st1) = pb1;
      __syncthreads();
    }
  }

  const int mw = m0 + wm * 32;
  if (MODE == 0 || MODE == 2) {
    for (int i = 0; i < 2; ++i) {
      int mb = mw + i * 16 + hi * 8;
      for (int j = 0; j < 4; ++j) {
        int col = n0 + wn * 64 + j * 16 + l15;
        for (int r = 0; r < 8; ++r) {
          int row = mb + r;
          float v = acc[i][j][r] + bias[row];
          if (MODE == 0) {
            unsigned short* o = (unsigned short*)Out + (size_t)b * CC * NPIX;
            o[(size_t)row * NPIX + col] = f32_to_bf16(v);
          } else {
            float* o = (float*)Out + (size_t)b * CC * NPIX;
            o[(size_t)row * NPIX + col] = v;
          }
        }
      }
    }
  } else {  // MODE 1: vt[b][h][n][d], d contiguous over accumulator rows
    for (int i = 0; i < 2; ++i) {
      int mb = mw + i * 16;
      int h = mb >> 6;
      int dbase = (mb & 63) + hi * 8;
      for (int j = 0; j < 4; ++j) {
        int n = n0 + wn * 64 + j * 16 + l15;
        Pack8 pk;
        for (int r2 = 0; r2 < 4; ++r2) {
          float v0 = acc[i][j][2 * r2] + bias[mb + hi * 8 + 2 * r2];
          float v1 = acc[i][j][2 * r2 + 1] + bias[mb + hi * 8 + 2 * r2 + 1];
          pk.w[r2] = (unsigned)f32_to_bf16(v0) | ((unsigned)f32_to_bf16(v1) << 16);
        }
        unsigned short* o =
            (unsigned short*)Out + (((size_t)(b * HN + h) * NPIX + n) * HC + dbase);
        *(v4u*)o = pk.u;
      }
    }
  }
}

// ---------------------------------------------------------------------------
// logits[bh][c][d] = scale * sum_n Q[bh][c][n] * K[bh][d][n]   (K-dim = 4096)
// one WG per (b,h); 8 waves * 2 tiles cover the 4x4 tile grid; frags from L2.
// ---------------------------------------------------------------------------
__global__ __launch_bounds__(256) void k_logits(const unsigned short* __restrict__ Q,
                                                const unsigned short* __restrict__ Kb,
                                                float* __restrict__ L) {
  const int bh = blockIdx.x;
  const int t = threadIdx.x, lane = t & 31, w = t >> 5;
  const int hi = (lane >> 4) & 1, l15 = lane & 15;
  const int m_tile = w & 3, dp = w >> 2;  // dp: 0..1 -> d tiles {2dp, 2dp+1}
  const unsigned short* qb = Q + (size_t)bh * HC * NPIX;   // rows are c within head
  const unsigned short* kb = Kb + (size_t)bh * HC * NPIX;
  const unsigned short* qrow = qb + (size_t)(m_tile * 16 + l15) * NPIX;
  const unsigned short* krow0 = kb + (size_t)((dp * 2 + 0) * 16 + l15) * NPIX;
  const unsigned short* krow1 = kb + (size_t)((dp * 2 + 1) * 16 + l15) * NPIX;

  v8f acc0 = {}, acc1 = {};
  for (int kk = 0; kk < NPIX; kk += 32) {
    FragU a, b0, b1;
    a.u[0] = *(const v4u*)(qrow + kk + hi * 16);
    a.u[1] = *(const v4u*)(qrow + kk + hi * 16 + 8);
    b0.u[0] = *(const v4u*)(krow0 + kk + hi * 16);
    b0.u[1] = *(const v4u*)(krow0 + kk + hi * 16 + 8);
    b1.u[0] = *(const v4u*)(krow1 + kk + hi * 16);
    b1.u[1] = *(const v4u*)(krow1 + kk + hi * 16 + 8);
    acc0 = wmma_bf16(a.bf, b0.bf, acc0);
    acc1 = wmma_bf16(a.bf, b1.bf, acc1);
  }
  float* lb = L + (size_t)bh * HC * HC;
  const float scale = 0.125f;  // hc^-0.5 = 1/8
  for (int r = 0; r < 8; ++r) {
    int crow = m_tile * 16 + hi * 8 + r;
    lb[crow * HC + (dp * 2 + 0) * 16 + l15] = acc0[r] * scale;
    lb[crow * HC + (dp * 2 + 1) * 16 + l15] = acc1[r] * scale;
  }
}

// ---------------------------------------------------------------------------
// softmax over last dim (64) of logits [4096 rows x 64] f32 -> bf16 probs
// one wave32 per row, 2 elements per lane
// ---------------------------------------------------------------------------
__global__ __launch_bounds__(256) void k_softmax(const float* __restrict__ L,
                                                 unsigned short* __restrict__ P) {
  const int lane = threadIdx.x & 31, wid = threadIdx.x >> 5;
  const int row = blockIdx.x * 8 + wid;
  const float* lr = L + (size_t)row * 64;
  float a0 = lr[lane], a1 = lr[lane + 32];
  float m = fmaxf(a0, a1);
  for (int off = 16; off; off >>= 1) m = fmaxf(m, __shfl_xor(m, off, 32));
  float e0 = __expf(a0 - m), e1 = __expf(a1 - m);
  float s = e0 + e1;
  for (int off = 16; off; off >>= 1) s += __shfl_xor(s, off, 32);
  float inv = 1.0f / s;
  P[(size_t)row * 64 + lane] = f32_to_bf16(e0 * inv);
  P[(size_t)row * 64 + lane + 32] = f32_to_bf16(e1 * inv);
}

// ---------------------------------------------------------------------------
// out2t[b][n][c] = sum_d P[bh][c][d] * VT[bh][n][d]   (K-dim = 64, 2 k-steps)
// grid (N/128, 64 bh); transposed pack-store so final GEMM reads contiguously
// ---------------------------------------------------------------------------
__global__ __launch_bounds__(256) void k_pv(const unsigned short* __restrict__ P,
                                            const unsigned short* __restrict__ VT,
                                            unsigned short* __restrict__ O2T) {
  const int bh = blockIdx.y, n0 = blockIdx.x * 128;
  const int t = threadIdx.x, lane = t & 31, w = t >> 5;
  const int hi = (lane >> 4) & 1, l15 = lane & 15;
  const int m_tile = w & 3, wn = w >> 2;
  const unsigned short* pb = P + (size_t)bh * HC * HC;
  const unsigned short* vb = VT + (size_t)bh * NPIX * HC;
  const unsigned short* prow = pb + (size_t)(m_tile * 16 + l15) * HC;

  v8f acc[4] = {};
  for (int kk = 0; kk < HC; kk += 32) {
    FragU a;
    a.u[0] = *(const v4u*)(prow + kk + hi * 16);
    a.u[1] = *(const v4u*)(prow + kk + hi * 16 + 8);
    for (int j = 0; j < 4; ++j) {
      int n = n0 + wn * 64 + j * 16 + l15;
      const unsigned short* vrow = vb + (size_t)n * HC + kk + hi * 16;
      FragU bf;
      bf.u[0] = *(const v4u*)(vrow);
      bf.u[1] = *(const v4u*)(vrow + 8);
      acc[j] = wmma_bf16(a.bf, bf.bf, acc[j]);
    }
  }
  const int b = bh >> 3, h = bh & 7;
  const int cbase = h * HC + m_tile * 16 + hi * 8;
  for (int j = 0; j < 4; ++j) {
    int n = n0 + wn * 64 + j * 16 + l15;
    Pack8 pk;
    for (int r2 = 0; r2 < 4; ++r2)
      pk.w[r2] = (unsigned)f32_to_bf16(acc[j][2 * r2]) |
                 ((unsigned)f32_to_bf16(acc[j][2 * r2 + 1]) << 16);
    *(v4u*)(O2T + ((size_t)b * NPIX + n) * CC + cbase) = pk.u;
  }
}

// ---------------------------------------------------------------------------
extern "C" void kernel_launch(void* const* d_in, const int* in_sizes, int n_in,
                              void* d_out, int out_size, void* d_ws, size_t ws_size,
                              hipStream_t stream) {
  (void)in_sizes; (void)n_in; (void)out_size; (void)ws_size;
  const float* x  = (const float*)d_in[0];
  const float* wq = (const float*)d_in[1];
  const float* bq = (const float*)d_in[2];
  const float* wk = (const float*)d_in[3];
  const float* bk = (const float*)d_in[4];
  const float* wv = (const float*)d_in[5];
  const float* bv = (const float*)d_in[6];
  const float* wp = (const float*)d_in[7];
  const float* bp = (const float*)d_in[8];
  float* out = (float*)d_out;

  char* ws = (char*)d_ws;
  size_t off = 0;
  auto alloc = [&](size_t bytes) -> void* {
    void* p = ws + off;
    off = (off + bytes + 255) & ~(size_t)255;
    return p;
  };
  unsigned short* XT  = (unsigned short*)alloc((size_t)NB * NPIX * CC * 2); // x^T bf16
  unsigned short* WQ  = (unsigned short*)alloc((size_t)CC * CC * 2);
  unsigned short* WK  = (unsigned short*)alloc((size_t)CC * CC * 2);
  unsigned short* WV  = (unsigned short*)alloc((size_t)CC * CC * 2);
  unsigned short* WP  = (unsigned short*)alloc((size_t)CC * CC * 2);
  unsigned short* Qb  = (unsigned short*)alloc((size_t)NB * CC * NPIX * 2); // [b][c][n]
  unsigned short* Kbf = (unsigned short*)alloc((size_t)NB * CC * NPIX * 2); // [b][c][n]
  unsigned short* VT  = (unsigned short*)alloc((size_t)NB * CC * NPIX * 2); // [bh][n][d]
  float*          LG  = (float*)alloc((size_t)NB * HN * HC * HC * 4);       // logits
  unsigned short* Pb  = (unsigned short*)alloc((size_t)NB * HN * HC * HC * 2);
  unsigned short* O2T = (unsigned short*)alloc((size_t)NB * NPIX * CC * 2); // [b][n][c]

  // 1. x -> transposed bf16
  k_transpose_cvt<<<dim3(NPIX / 32, CC / 32, NB), dim3(32, 8), 0, stream>>>(x, XT);
  // 2. weights -> bf16
  k_cvt<<<(CC * CC + 255) / 256, 256, 0, stream>>>(wq, WQ, CC * CC);
  k_cvt<<<(CC * CC + 255) / 256, 256, 0, stream>>>(wk, WK, CC * CC);
  k_cvt<<<(CC * CC + 255) / 256, 256, 0, stream>>>(wv, WV, CC * CC);
  k_cvt<<<(CC * CC + 255) / 256, 256, 0, stream>>>(wp, WP, CC * CC);
  // 3. Q/K/V projections (V stored transposed per head)
  dim3 ggrid(NPIX / 128, CC / 128, NB);
  k_gemm<0><<<ggrid, 256, 0, stream>>>(WQ, XT, bq, (void*)Qb);
  k_gemm<0><<<ggrid, 256, 0, stream>>>(WK, XT, bk, (void*)Kbf);
  k_gemm<1><<<ggrid, 256, 0, stream>>>(WV, XT, bv, (void*)VT);
  // 4. attention logits + softmax
  k_logits<<<NB * HN, 256, 0, stream>>>(Qb, Kbf, LG);
  k_softmax<<<(NB * HN * HC) / 8, 256, 0, stream>>>(LG, Pb);
  // 5. P @ V -> transposed attention output
  k_pv<<<dim3(NPIX / 128, NB * HN), 256, 0, stream>>>(Pb, VT, O2T);
  // 6. final projection -> f32 output
  k_gemm<2><<<ggrid, 256, 0, stream>>>(WP, O2T, bp, (void*)out);
}
